// _DSTFormerV5GuidedMoEBlock_43061342110338
// MI455X (gfx1250) — matrix operand order repaired and checked by reading, not compiled, not run on hardware
//
#include <hip/hip_runtime.h>
#include <hip/hip_bf16.h>
#include <cstddef>

// ---------------------------------------------------------------------------
// DSTFormer dual-stream block for MI455X (gfx1250, wave32, WMMA).
// Heavy GEMMs run on v_wmma_f32_16x16x32_bf16 with *pre-converted* bf16
// operands: fragment loads go straight into fragment registers (full MLP,
// no staging-register serialization, no in-loop converts), fp32 accumulate.
// K is a multiple of 16 everywhere (MoE hidden dim padded 682->688 with
// zero-fill), so WMMA inner loops are pure: 10/18x global_load_b128 + 4/8 wmma.
// ---------------------------------------------------------------------------

// ---------------- problem constants ----------------
constexpr int Bc    = 4;
constexpr int Tc    = 243;
constexpr int Nc    = 17;
constexpr int DIMc  = 256;
constexpr int HEADS = 8;
constexpr int HDc   = 32;          // head dim
constexpr int Ec    = 8;           // experts
constexpr int HIDc  = 682;         // int(256*4*2/3)
constexpr int HPAD  = 688;         // HID padded to multiple of 16 (zero-filled)
constexpr int MTOK  = Bc * Tc * Nc;   // 16524 tokens
constexpr int MG    = Bc * Tc;        // 972 guide rows
constexpr int G_SP  = Bc * Tc;        // spatial attention groups (S = N = 17)
constexpr int S_SP  = Nc;
constexpr int G_TM  = Bc * Nc;        // temporal attention groups (S = T = 243)
constexpr int S_TM  = Tc;

// ---------------- WMMA types & helpers ----------------
typedef __bf16 v16bf __attribute__((ext_vector_type(16)));
typedef __bf16 v8bf  __attribute__((ext_vector_type(8)));
typedef float  v8f   __attribute__((ext_vector_type(8)));
// under-aligned views for loads (backend emits align-4 b128, fine on gfx1250)
typedef __bf16 v16bf_u __attribute__((ext_vector_type(16), aligned(4)));
typedef __bf16 v8bf_u  __attribute__((ext_vector_type(8),  aligned(4)));

__device__ __forceinline__ v8f v8f_zero() {
  v8f z;
#pragma unroll
  for (int i = 0; i < 8; ++i) z[i] = 0.f;
  return z;
}

__device__ __forceinline__ v8f wmma_bf16(v16bf a, v16bf b, v8f c) {
  // D = A(16x32 bf16) * B(32x16 bf16) + C(16x16 f32)
  return __builtin_amdgcn_wmma_f32_16x16x32_bf16(
      /*neg_a=*/false, a, /*neg_b=*/false, b,
      /*c_mod=*/(short)0, c, /*reuse_a=*/false, /*reuse_b=*/false);
}

__device__ __forceinline__ int imin(int a, int b) { return a < b ? a : b; }

// A fragment: p = &A[row*K + kg + k0] (bf16). Contiguous K order:
// lanes 0-15 hold row, K = {0..15}; lanes 16-31 hold K = {16..31}.
// Two contiguous 8-bf16 (16B) chunks at p[0..7] and p[8..15].
__device__ __forceinline__ v16bf load_a_frag(const __bf16* __restrict__ p) {
  const v8bf_u lo = *(const v8bf_u*)(p);
  const v8bf_u hi = *(const v8bf_u*)(p + 8);
  v16bf a;
#pragma unroll
  for (int i = 0; i < 8; ++i) { a[i] = lo[i]; a[i + 8] = hi[i]; }
  return a;
}

// B fragment: B = W^T, W row-major [N,K] bf16. p = &W[col*K + kh + k0];
// lanes 0-15 hold col, K contiguous (kh = 0 / 16 per lane half). 32B load.
__device__ __forceinline__ v16bf load_b_frag(const __bf16* __restrict__ p) {
  const v16bf_u t = *(const v16bf_u*)p;
  v16bf b;
#pragma unroll
  for (int i = 0; i < 16; ++i) b[i] = t[i];
  return b;
}

// ---------------- GEMM: C[M,N] = A[M,K] @ W[N,K]^T + bias[N] ----------------
// bf16 inputs, fp32 output. K must be a multiple of 32 (true for all uses).
// Wave computes a 16x64 tile: 4 x v_wmma_f32_16x16x32_bf16 per 32-wide K-step.
__global__ __launch_bounds__(256) void gemm_bf16_kernel(
    const __bf16* __restrict__ A, const __bf16* __restrict__ W,
    const float* __restrict__ bias, float* __restrict__ C,
    int M, int Nn, int K) {
  const int lane = threadIdx.x & 31;
  const int wid  = threadIdx.x >> 5;
  const int ntile = (Nn + 63) >> 6;
  const int mtile = (M + 15) >> 4;
  const int tile = blockIdx.x * 8 + wid;
  if (tile >= ntile * mtile) return;        // wave-uniform
  const int mt = tile / ntile;
  const int nt = tile - mt * ntile;
  const int row0 = mt << 4;
  const int n0 = nt << 6;

  const int kg = (lane >> 4) << 4;          // A K-offset: 0 or 16
  const int kh = (lane >> 4) << 4;          // B K-offset: 0 or 16
  const __bf16* pa =
      A + (size_t)((unsigned)imin(row0 + (lane & 15), M - 1) * (unsigned)K + kg);
  const __bf16* pb0 =
      W + (size_t)((unsigned)imin(n0 +  0 + (lane & 15), Nn - 1) * (unsigned)K + kh);
  const __bf16* pb1 =
      W + (size_t)((unsigned)imin(n0 + 16 + (lane & 15), Nn - 1) * (unsigned)K + kh);
  const __bf16* pb2 =
      W + (size_t)((unsigned)imin(n0 + 32 + (lane & 15), Nn - 1) * (unsigned)K + kh);
  const __bf16* pb3 =
      W + (size_t)((unsigned)imin(n0 + 48 + (lane & 15), Nn - 1) * (unsigned)K + kh);

  v8f acc0 = v8f_zero(), acc1 = v8f_zero(), acc2 = v8f_zero(), acc3 = v8f_zero();

  for (int k0 = 0; k0 < K; k0 += 32) {      // branchless hot loop
    const v16bf a  = load_a_frag(pa + k0);
    const v16bf b0 = load_b_frag(pb0 + k0);
    const v16bf b1 = load_b_frag(pb1 + k0);
    const v16bf b2 = load_b_frag(pb2 + k0);
    const v16bf b3 = load_b_frag(pb3 + k0);
    acc0 = wmma_bf16(a, b0, acc0);
    acc1 = wmma_bf16(a, b1, acc1);
    acc2 = wmma_bf16(a, b2, acc2);
    acc3 = wmma_bf16(a, b3, acc3);
  }

  // C layout: acc[r] -> row = row0 + r + 8*(lane>>4), col = n0sub + (lane&15)
  const int rbase = row0 + ((lane >> 4) << 3);
  const v8f* accs[4] = {&acc0, &acc1, &acc2, &acc3};
#pragma unroll
  for (int j = 0; j < 4; ++j) {
    const int col = n0 + j * 16 + (lane & 15);
    if (col < Nn) {
      const float bv = bias[col];
      const v8f av = *accs[j];
#pragma unroll
      for (int r = 0; r < 8; ++r) {
        const int row = rbase + r;
        if (row < M) C[(size_t)row * Nn + col] = av[r] + bv;
      }
    }
  }
}

// ---------------- MoE expert hidden: H = silu(A@W1^T+b1) * (A@W3^T+b3) -----
// bf16 in, bf16 out with ld = HPAD; padded cols [Nn, HPAD) written as zero.
__global__ __launch_bounds__(256) void moe_h_kernel(
    const __bf16* __restrict__ A, const __bf16* __restrict__ W1,
    const __bf16* __restrict__ W3, const float* __restrict__ b1,
    const float* __restrict__ b3, __bf16* __restrict__ H,
    int M, int Nn, int K) {
  const int lane = threadIdx.x & 31;
  const int wid  = threadIdx.x >> 5;
  const int ntile = (Nn + 63) >> 6;
  const int mtile = (M + 15) >> 4;
  const int tile = blockIdx.x * 8 + wid;
  if (tile >= ntile * mtile) return;
  const int mt = tile / ntile;
  const int nt = tile - mt * ntile;
  const int row0 = mt << 4;
  const int n0 = nt << 6;

  const int kg = (lane >> 4) << 4;
  const int kh = (lane >> 4) << 4;
  const __bf16* pa =
      A + (size_t)((unsigned)imin(row0 + (lane & 15), M - 1) * (unsigned)K + kg);
  // same column offsets apply to W1 and W3 (both [Nn,K] row-major)
  const unsigned off0 = (unsigned)imin(n0 +  0 + (lane & 15), Nn - 1) * (unsigned)K + kh;
  const unsigned off1 = (unsigned)imin(n0 + 16 + (lane & 15), Nn - 1) * (unsigned)K + kh;
  const unsigned off2 = (unsigned)imin(n0 + 32 + (lane & 15), Nn - 1) * (unsigned)K + kh;
  const unsigned off3 = (unsigned)imin(n0 + 48 + (lane & 15), Nn - 1) * (unsigned)K + kh;

  v8f g0 = v8f_zero(), g1 = v8f_zero(), g2 = v8f_zero(), g3 = v8f_zero();
  v8f u0 = v8f_zero(), u1 = v8f_zero(), u2 = v8f_zero(), u3 = v8f_zero();

  for (int k0 = 0; k0 < K; k0 += 32) {
    const v16bf a   = load_a_frag(pa + k0);
    const v16bf w10 = load_b_frag(W1 + off0 + k0);
    const v16bf w11 = load_b_frag(W1 + off1 + k0);
    const v16bf w12 = load_b_frag(W1 + off2 + k0);
    const v16bf w13 = load_b_frag(W1 + off3 + k0);
    const v16bf w30 = load_b_frag(W3 + off0 + k0);
    const v16bf w31 = load_b_frag(W3 + off1 + k0);
    const v16bf w32 = load_b_frag(W3 + off2 + k0);
    const v16bf w33 = load_b_frag(W3 + off3 + k0);
    g0 = wmma_bf16(a, w10, g0);
    g1 = wmma_bf16(a, w11, g1);
    g2 = wmma_bf16(a, w12, g2);
    g3 = wmma_bf16(a, w13, g3);
    u0 = wmma_bf16(a, w30, u0);
    u1 = wmma_bf16(a, w31, u1);
    u2 = wmma_bf16(a, w32, u2);
    u3 = wmma_bf16(a, w33, u3);
  }

  const int rbase = row0 + ((lane >> 4) << 3);
  const v8f* gg[4] = {&g0, &g1, &g2, &g3};
  const v8f* uu[4] = {&u0, &u1, &u2, &u3};
#pragma unroll
  for (int j = 0; j < 4; ++j) {
    const int col = n0 + j * 16 + (lane & 15);
    if (col < HPAD) {
      const bool live = (col < Nn);
      const float bg = live ? b1[col] : 0.f;
      const float bu = live ? b3[col] : 0.f;
      const v8f gv = *gg[j];
      const v8f uv = *uu[j];
#pragma unroll
      for (int r = 0; r < 8; ++r) {
        const int row = rbase + r;
        if (row < M) {
          const float g = gv[r] + bg;
          const float u = uv[r] + bu;
          const float s = g / (1.f + __expf(-g));   // silu
          H[(size_t)row * HPAD + col] = live ? (__bf16)(s * u) : (__bf16)0.f;
        }
      }
    }
  }
}

// ---------------- MoE down-proj accumulate: out += rw * (H@W2^T + b2) ------
// H: bf16 [M, HPAD]; W2: bf16 [Nn, HPAD] (padded, zero-filled); K = HPAD.
__global__ __launch_bounds__(256) void moe_w2_kernel(
    const __bf16* __restrict__ H, const __bf16* __restrict__ W2,
    const float* __restrict__ b2, const float* __restrict__ rwe,  // stride E
    float* __restrict__ out, int M, int Nn, int K) {
  const int lane = threadIdx.x & 31;
  const int wid  = threadIdx.x >> 5;
  const int ntile = (Nn + 63) >> 6;
  const int mtile = (M + 15) >> 4;
  const int tile = blockIdx.x * 8 + wid;
  if (tile >= ntile * mtile) return;
  const int mt = tile / ntile;
  const int nt = tile - mt * ntile;
  const int row0 = mt << 4;
  const int n0 = nt << 6;

  const int kg = (lane >> 4) << 4;
  const int kh = (lane >> 4) << 4;
  const __bf16* pa =
      H + (size_t)((unsigned)imin(row0 + (lane & 15), M - 1) * (unsigned)K + kg);
  const __bf16* pb0 =
      W2 + (size_t)((unsigned)imin(n0 +  0 + (lane & 15), Nn - 1) * (unsigned)K + kh);
  const __bf16* pb1 =
      W2 + (size_t)((unsigned)imin(n0 + 16 + (lane & 15), Nn - 1) * (unsigned)K + kh);
  const __bf16* pb2 =
      W2 + (size_t)((unsigned)imin(n0 + 32 + (lane & 15), Nn - 1) * (unsigned)K + kh);
  const __bf16* pb3 =
      W2 + (size_t)((unsigned)imin(n0 + 48 + (lane & 15), Nn - 1) * (unsigned)K + kh);

  v8f acc0 = v8f_zero(), acc1 = v8f_zero(), acc2 = v8f_zero(), acc3 = v8f_zero();

  for (int k0 = 0; k0 < K; k0 += 32) {
    const v16bf a  = load_a_frag(pa + k0);
    const v16bf b0 = load_b_frag(pb0 + k0);
    const v16bf b1 = load_b_frag(pb1 + k0);
    const v16bf b2v = load_b_frag(pb2 + k0);
    const v16bf b3v = load_b_frag(pb3 + k0);
    acc0 = wmma_bf16(a, b0, acc0);
    acc1 = wmma_bf16(a, b1, acc1);
    acc2 = wmma_bf16(a, b2v, acc2);
    acc3 = wmma_bf16(a, b3v, acc3);
  }

  const int rbase = row0 + ((lane >> 4) << 3);
  const v8f* accs[4] = {&acc0, &acc1, &acc2, &acc3};
#pragma unroll
  for (int j = 0; j < 4; ++j) {
    const int col = n0 + j * 16 + (lane & 15);
    if (col < Nn) {
      const float bv = b2[col];
      const v8f av = *accs[j];
#pragma unroll
      for (int r = 0; r < 8; ++r) {
        const int row = rbase + r;
        if (row < M) {
          const float w = rwe[(size_t)row * Ec];
          out[(size_t)row * Nn + col] += w * (av[r] + bv);
        }
      }
    }
  }
}

// ---------------- fp32 -> bf16 conversions ----------------
__global__ void cvt_bf16_kernel(const float* __restrict__ in,
                                __bf16* __restrict__ out, size_t n) {
  const size_t i = (size_t)blockIdx.x * blockDim.x + threadIdx.x;
  if (i < n) out[i] = (__bf16)in[i];
}

// rows x ldin fp32 -> rows x ldout bf16 with zero-fill in [ldin, ldout)
__global__ void cvt_pad_kernel(const float* __restrict__ in,
                               __bf16* __restrict__ out,
                               int rows, int ldin, int ldout) {
  const size_t i = (size_t)blockIdx.x * blockDim.x + threadIdx.x;
  if (i >= (size_t)rows * ldout) return;
  const int r = (int)(i / ldout);
  const int c = (int)(i - (size_t)r * ldout);
  out[i] = (c < ldin) ? (__bf16)in[(size_t)r * ldin + c] : (__bf16)0.f;
}

// ---------------- LayerNorm -> bf16, optional (b,t,n)->(b,n,t) gather ------
// mode 0: out[row] = LN(x[row]); mode 1: out rows in temporal order.
__global__ __launch_bounds__(256) void ln_kernel(
    const float* __restrict__ x, const float* __restrict__ gamma,
    const float* __restrict__ beta, __bf16* __restrict__ out, int mode) {
  const int lane = threadIdx.x & 31;
  const int wave = threadIdx.x >> 5;
  const int orow = blockIdx.x * 8 + wave;
  if (orow >= MTOK) return;
  int irow = orow;
  if (mode) {
    const int bb = orow / (Nc * Tc);
    const int r  = orow % (Nc * Tc);
    const int n  = r / Tc;
    const int t  = r - n * Tc;
    irow = (bb * Tc + t) * Nc + n;
  }
  const float* xr = x + (size_t)irow * DIMc;
  float v[8];
  float s = 0.f;
#pragma unroll
  for (int i = 0; i < 8; ++i) { v[i] = xr[lane * 8 + i]; s += v[i]; }
#pragma unroll
  for (int off = 16; off > 0; off >>= 1) s += __shfl_xor(s, off, 32);
  const float mean = s * (1.f / DIMc);
  float vs = 0.f;
#pragma unroll
  for (int i = 0; i < 8; ++i) { const float d = v[i] - mean; vs += d * d; }
#pragma unroll
  for (int off = 16; off > 0; off >>= 1) vs += __shfl_xor(vs, off, 32);
  const float inv = rsqrtf(vs * (1.f / DIMc) + 1e-5f);
  __bf16* orr = out + (size_t)orow * DIMc;
#pragma unroll
  for (int i = 0; i < 8; ++i) {
    const int c = lane * 8 + i;
    orr[c] = (__bf16)((v[i] - mean) * inv * gamma[c] + beta[c]);
  }
}

// ---------------- residual adds ----------------
__global__ void add_kernel(float* __restrict__ x, const float* __restrict__ y,
                           size_t n) {
  const size_t i = (size_t)blockIdx.x * blockDim.x + threadIdx.x;
  if (i < n) x[i] += y[i];
}

// y rows are in temporal (b,n,t) order; scatter-add back to spatial x.
__global__ void add_t_kernel(float* __restrict__ x, const float* __restrict__ y) {
  const size_t i = (size_t)blockIdx.x * blockDim.x + threadIdx.x;
  if (i >= (size_t)MTOK * DIMc) return;
  const int ot = (int)(i >> 8);
  const int c  = (int)(i & 255);
  const int bb = ot / (Nc * Tc);
  const int r  = ot % (Nc * Tc);
  const int n  = r / Tc;
  const int t  = r - n * Tc;
  const int ir = (bb * Tc + t) * Nc + n;
  x[(size_t)ir * DIMc + c] += y[i];
}

__global__ void zero_kernel(float* __restrict__ p, size_t n) {
  const size_t i = (size_t)blockIdx.x * blockDim.x + threadIdx.x;
  if (i < n) p[i] = 0.f;
}

// ---------------- guide bias: gb[MG,E] = guide @ gw^T + gbias --------------
__global__ void gbias_kernel(const float* __restrict__ guide,
                             const float* __restrict__ gw,
                             const float* __restrict__ gbias,
                             float* __restrict__ gb) {
  const int idx = blockIdx.x * blockDim.x + threadIdx.x;
  if (idx >= MG * Ec) return;
  const int g = idx / Ec;
  const int e = idx - g * Ec;
  const float* xr = guide + (size_t)g * DIMc;
  const float* wr = gw + (size_t)e * DIMc;
  float d = 0.f;
  for (int c = 0; c < DIMc; ++c) d += xr[c] * wr[c];
  gb[idx] = d + gbias[e];
}

// ---------------- router: rw[m,:] = softmax(x@W^T + b + gb[map(m)]) --------
__global__ void router_kernel(const __bf16* __restrict__ xn,
                              const float* __restrict__ rw_w,
                              const float* __restrict__ rw_b,
                              const float* __restrict__ gb,
                              float* __restrict__ out, int mode) {
  const int m = blockIdx.x * blockDim.x + threadIdx.x;
  if (m >= MTOK) return;
  int grow;
  if (mode == 0) grow = m / Nc;                       // spatial: row = (bt)*N+n
  else {                                              // temporal: row = (bN+n)*T+t
    const int bb = m / (Nc * Tc);
    const int t  = m % Tc;
    grow = bb * Tc + t;
  }
  const __bf16* xr = xn + (size_t)m * DIMc;
  float lg[Ec];
  float mx = -1e30f;
#pragma unroll
  for (int e = 0; e < Ec; ++e) {
    const float* wr = rw_w + (size_t)e * DIMc;
    float d = 0.f;
    for (int c = 0; c < DIMc; ++c) d += (float)xr[c] * wr[c];
    lg[e] = d + rw_b[e] + gb[(size_t)grow * Ec + e];
    mx = fmaxf(mx, lg[e]);
  }
  float den = 0.f;
#pragma unroll
  for (int e = 0; e < Ec; ++e) { lg[e] = __expf(lg[e] - mx); den += lg[e]; }
  const float inv = 1.f / den;
#pragma unroll
  for (int e = 0; e < Ec; ++e) out[(size_t)m * Ec + e] = lg[e] * inv;
}

// ---------------- fused attention (online softmax, K/V in LDS) -------------
// qkv rows: [q(8x32) | k(8x32) | v(8x32)] fp32. Output written as bf16
// (feeds the projection GEMM directly). One block per (group, head).
__global__ __launch_bounds__(256) void attn_kernel(
    const float* __restrict__ qkv, __bf16* __restrict__ o, int S,
    const float* __restrict__ fcos, const float* __restrict__ fsin, int rope) {
  __shared__ float ks[S_TM * HDc];
  __shared__ float vs[S_TM * HDc];
  const int g = blockIdx.x;
  const int h = blockIdx.y;
  const float* base = qkv + (size_t)g * S * (3 * DIMc);

  for (int idx = threadIdx.x; idx < S * (HDc / 2); idx += blockDim.x) {
    const int s = idx >> 4;          // HD/2 = 16 pairs
    const int i = idx & 15;
    const float* kr = base + (size_t)s * (3 * DIMc) + DIMc + h * HDc;
    const float* vr = kr + DIMc;
    float k0 = kr[2 * i], k1 = kr[2 * i + 1];
    if (rope) {
      const float c = fcos[s * 16 + i], sn = fsin[s * 16 + i];
      const float t0 = k0 * c - k1 * sn;
      const float t1 = k0 * sn + k1 * c;
      k0 = t0; k1 = t1;
    }
    ks[s * HDc + 2 * i]     = k0;
    ks[s * HDc + 2 * i + 1] = k1;
    vs[s * HDc + 2 * i]     = vr[2 * i];
    vs[s * HDc + 2 * i + 1] = vr[2 * i + 1];
  }
  __syncthreads();

  const int q = threadIdx.x;
  if (q < S) {
    float qv[HDc];
    const float* qr = base + (size_t)q * (3 * DIMc) + h * HDc;
#pragma unroll
    for (int i = 0; i < 16; ++i) {
      float q0 = qr[2 * i], q1 = qr[2 * i + 1];
      if (rope) {
        const float c = fcos[q * 16 + i], sn = fsin[q * 16 + i];
        const float t0 = q0 * c - q1 * sn;
        const float t1 = q0 * sn + q1 * c;
        q0 = t0; q1 = t1;
      }
      qv[2 * i] = q0; qv[2 * i + 1] = q1;
    }
    float mmax = -1e30f, l = 0.f, acc[HDc];
#pragma unroll
    for (int i = 0; i < HDc; ++i) acc[i] = 0.f;
    for (int s2 = 0; s2 < S; ++s2) {
      float d = 0.f;
#pragma unroll
      for (int i = 0; i < HDc; ++i) d += qv[i] * ks[s2 * HDc + i];
      d *= 0.17677669529663687f;     // 1/sqrt(32)
      const float nm = fmaxf(mmax, d);
      const float corr = __expf(mmax - nm);
      const float p = __expf(d - nm);
      l = l * corr + p;
#pragma unroll
      for (int i = 0; i < HDc; ++i)
        acc[i] = acc[i] * corr + p * vs[s2 * HDc + i];
      mmax = nm;
    }
    const float inv = 1.f / l;
    __bf16* orow = o + ((size_t)g * S + q) * DIMc + h * HDc;
#pragma unroll
    for (int i = 0; i < HDc; ++i) orow[i] = (__bf16)(acc[i] * inv);
  }
}

// ---------------- stream fusion ----------------
// alpha = softmax([x_st,x_ts] @ Wf^T + bf); out = x_st*a0 + x_ts*a1
__global__ void fuse_kernel(const float* __restrict__ xst,
                            const float* __restrict__ xts,
                            const float* __restrict__ w,
                            const float* __restrict__ b,
                            float* __restrict__ out) {
  const int m = blockIdx.x * blockDim.x + threadIdx.x;
  if (m >= MTOK) return;
  const float* a = xst + (size_t)m * DIMc;
  const float* c = xts + (size_t)m * DIMc;
  float l0 = b[0], l1 = b[1];
  for (int i = 0; i < DIMc; ++i) {
    l0 += a[i] * w[i] + c[i] * w[DIMc + i];
    l1 += a[i] * w[512 + i] + c[i] * w[512 + DIMc + i];
  }
  const float mx = fmaxf(l0, l1);
  const float e0 = __expf(l0 - mx), e1 = __expf(l1 - mx);
  const float a0 = e0 / (e0 + e1);
  const float a1 = 1.f - a0;
  float* orow = out + (size_t)m * DIMc;
  for (int i = 0; i < DIMc; ++i) orow[i] = a[i] * a0 + c[i] * a1;
}

// ===========================================================================
// Host orchestration
// ===========================================================================
static inline int ceil_div(int a, int b) { return (a + b - 1) / b; }
static inline size_t cdz(size_t a, size_t b) { return (a + b - 1) / b; }

struct MoeP {
  const float *b1, *b2, *b3, *gb, *gw, *rb, *rw, *w1, *w2, *w3;
};

static void launch_gemm(const __bf16* A, const __bf16* W, const float* b,
                        float* C, int M, int Nn, int K, hipStream_t s) {
  const int tiles = ceil_div(M, 16) * ceil_div(Nn, 64);
  gemm_bf16_kernel<<<ceil_div(tiles, 8), 256, 0, s>>>(A, W, b, C, M, Nn, K);
}

static void cvt(const float* in, __bf16* out, size_t n, hipStream_t s) {
  cvt_bf16_kernel<<<(int)cdz(n, 256), 256, 0, s>>>(in, out, n);
}

static void run_moe(const MoeP& p, int mode, const __bf16* xn,
                    const float* guide, __bf16* H, __bf16* wb16,
                    float* rwb, float* gbb, float* out, hipStream_t s) {
  constexpr size_t W1SZ = (size_t)Ec * HIDc * DIMc;
  __bf16* w1_16 = wb16;
  __bf16* w3_16 = wb16 + W1SZ;
  __bf16* w2_16 = wb16 + 2 * W1SZ;            // [Ec*DIMc, HPAD] zero-padded
  cvt(p.w1, w1_16, W1SZ, s);
  cvt(p.w3, w3_16, W1SZ, s);
  cvt_pad_kernel<<<(int)cdz((size_t)Ec * DIMc * HPAD, 256), 256, 0, s>>>(
      p.w2, w2_16, Ec * DIMc, HIDc, HPAD);
  gbias_kernel<<<ceil_div(MG * Ec, 128), 128, 0, s>>>(guide, p.gw, p.gb, gbb);
  router_kernel<<<ceil_div(MTOK, 128), 128, 0, s>>>(xn, p.rw, p.rb, gbb, rwb, mode);
  zero_kernel<<<ceil_div(MTOK * DIMc, 256), 256, 0, s>>>(out, (size_t)MTOK * DIMc);
  const int tiles_h  = ceil_div(MTOK, 16) * ceil_div(HIDc, 64);
  const int tiles_w2 = ceil_div(MTOK, 16) * ceil_div(DIMc, 64);
  for (int e = 0; e < Ec; ++e) {
    moe_h_kernel<<<ceil_div(tiles_h, 8), 256, 0, s>>>(
        xn, w1_16 + (size_t)e * HIDc * DIMc, w3_16 + (size_t)e * HIDc * DIMc,
        p.b1 + (size_t)e * HIDc, p.b3 + (size_t)e * HIDc, H, MTOK, HIDc, DIMc);
    moe_w2_kernel<<<ceil_div(tiles_w2, 8), 256, 0, s>>>(
        H, w2_16 + (size_t)e * DIMc * HPAD, p.b2 + (size_t)e * DIMc,
        rwb + e, out, MTOK, DIMc, HPAD);
  }
}

extern "C" void kernel_launch(void* const* d_in, const int* in_sizes, int n_in,
                              void* d_out, int out_size, void* d_ws, size_t ws_size,
                              hipStream_t stream) {
  (void)in_sizes; (void)n_in; (void)out_size; (void)ws_size;
  auto F = [&](int i) { return (const float*)d_in[i]; };

  // Leaf order: jax tree-flatten of setup_inputs() dict (keys sorted at every
  // level): freqs_cos, freqs_sin, guide, params{...}, x.
  const float* fcos  = F(0);
  const float* fsin  = F(1);
  const float* guide = F(2);
  const float* fuse_b = F(3);
  const float* fuse_w = F(4);
  // st_s_attn: out{b,w}, qkv{b,w}
  const float* sts_out_b = F(5);  const float* sts_out_w = F(6);
  const float* sts_qkv_b = F(7);  const float* sts_qkv_w = F(8);
  auto moe_at = [&](int i) {
    return MoeP{F(i), F(i + 1), F(i + 2), F(i + 3), F(i + 4),
                F(i + 5), F(i + 6), F(i + 7), F(i + 8), F(i + 9)};
  };
  const MoeP st_s_ffn = moe_at(9);    // b1,b2,b3,guide.b,guide.w,router.b,router.w,w1,w2,w3
  const float* sts_ffn_ln_b = F(19); const float* sts_ffn_ln_g = F(20);
  const float* sts_ln_b     = F(21); const float* sts_ln_g     = F(22);
  const MoeP st_t_ffn = moe_at(23);
  const float* stt_ffn_ln_b = F(33); const float* stt_ffn_ln_g = F(34);
  const float* stt_ln_b     = F(35); const float* stt_ln_g     = F(36);
  const float* stt_proj_b = F(37); const float* stt_proj_w = F(38);
  const float* stt_qkv_b  = F(39); const float* stt_qkv_w  = F(40);
  const float* tss_out_b = F(41); const float* tss_out_w = F(42);
  const float* tss_qkv_b = F(43); const float* tss_qkv_w = F(44);
  const MoeP ts_s_ffn = moe_at(45);
  const float* tss_ffn_ln_b = F(55); const float* tss_ffn_ln_g = F(56);
  const float* tss_ln_b     = F(57); const float* tss_ln_g     = F(58);
  const MoeP ts_t_ffn = moe_at(59);
  const float* tst_ffn_ln_b = F(69); const float* tst_ffn_ln_g = F(70);
  const float* tst_ln_b     = F(71); const float* tst_ln_g     = F(72);
  const float* tst_proj_b = F(73); const float* tst_proj_w = F(74);
  const float* tst_qkv_b  = F(75); const float* tst_qkv_w  = F(76);
  const float* x = F(77);

  // ---- workspace carve (~150 MB) ----
  float* p = (float*)d_ws;
  float* xst   = p; p += (size_t)MTOK * DIMc;
  float* xts   = p; p += (size_t)MTOK * DIMc;
  float* qkvh  = p; p += (size_t)MTOK * 3 * DIMc;
  float* ybuf  = p; p += (size_t)MTOK * DIMc;
  float* rwb   = p; p += (size_t)MTOK * Ec;
  float* gbb   = p; p += (size_t)MG * Ec;
  __bf16* mb16 = (__bf16*)p;                 // LN output / attention output
  __bf16* h16  = mb16 + (size_t)MTOK * DIMc; // MoE hidden [MTOK, HPAD]
  __bf16* wb16 = h16 + (size_t)MTOK * HPAD;  // converted weights (2*W1SZ + W2P)

  const size_t xbytes = (size_t)MTOK * DIMc * sizeof(float);
  hipMemcpyAsync(xst, x, xbytes, hipMemcpyDeviceToDevice, stream);
  hipMemcpyAsync(xts, x, xbytes, hipMemcpyDeviceToDevice, stream);

  const int lnb = ceil_div(MTOK, 8);
  const size_t ntot = (size_t)MTOK * DIMc;
  const int addb = (int)cdz(ntot, 256);
  const size_t QKVW = (size_t)(3 * DIMc) * DIMc;
  const size_t PRJW = (size_t)DIMc * DIMc;

  // ================= ST stream =================
  // spatial attention
  ln_kernel<<<lnb, 256, 0, stream>>>(xst, sts_ln_g, sts_ln_b, mb16, 0);
  cvt(sts_qkv_w, wb16, QKVW, stream);
  launch_gemm(mb16, wb16, sts_qkv_b, qkvh, MTOK, 3 * DIMc, DIMc, stream);
  attn_kernel<<<dim3(G_SP, HEADS), 256, 0, stream>>>(qkvh, mb16, S_SP, fcos, fsin, 0);
  cvt(sts_out_w, wb16, PRJW, stream);
  launch_gemm(mb16, wb16, sts_out_b, ybuf, MTOK, DIMc, DIMc, stream);
  add_kernel<<<addb, 256, 0, stream>>>(xst, ybuf, ntot);
  // spatial MoE
  ln_kernel<<<lnb, 256, 0, stream>>>(xst, sts_ffn_ln_g, sts_ffn_ln_b, mb16, 0);
  run_moe(st_s_ffn, 0, mb16, guide, h16, wb16, rwb, gbb, ybuf, stream);
  add_kernel<<<addb, 256, 0, stream>>>(xst, ybuf, ntot);
  // temporal attention (RoPE)
  ln_kernel<<<lnb, 256, 0, stream>>>(xst, stt_ln_g, stt_ln_b, mb16, 1);
  cvt(stt_qkv_w, wb16, QKVW, stream);
  launch_gemm(mb16, wb16, stt_qkv_b, qkvh, MTOK, 3 * DIMc, DIMc, stream);
  attn_kernel<<<dim3(G_TM, HEADS), 256, 0, stream>>>(qkvh, mb16, S_TM, fcos, fsin, 1);
  cvt(stt_proj_w, wb16, PRJW, stream);
  launch_gemm(mb16, wb16, stt_proj_b, ybuf, MTOK, DIMc, DIMc, stream);
  add_t_kernel<<<addb, 256, 0, stream>>>(xst, ybuf);
  // temporal MoE
  ln_kernel<<<lnb, 256, 0, stream>>>(xst, stt_ffn_ln_g, stt_ffn_ln_b, mb16, 1);
  run_moe(st_t_ffn, 1, mb16, guide, h16, wb16, rwb, gbb, ybuf, stream);
  add_t_kernel<<<addb, 256, 0, stream>>>(xst, ybuf);

  // ================= TS stream =================
  // temporal attention (RoPE)
  ln_kernel<<<lnb, 256, 0, stream>>>(xts, tst_ln_g, tst_ln_b, mb16, 1);
  cvt(tst_qkv_w, wb16, QKVW, stream);
  launch_gemm(mb16, wb16, tst_qkv_b, qkvh, MTOK, 3 * DIMc, DIMc, stream);
  attn_kernel<<<dim3(G_TM, HEADS), 256, 0, stream>>>(qkvh, mb16, S_TM, fcos, fsin, 1);
  cvt(tst_proj_w, wb16, PRJW, stream);
  launch_gemm(mb16, wb16, tst_proj_b, ybuf, MTOK, DIMc, DIMc, stream);
  add_t_kernel<<<addb, 256, 0, stream>>>(xts, ybuf);
  // temporal MoE
  ln_kernel<<<lnb, 256, 0, stream>>>(xts, tst_ffn_ln_g, tst_ffn_ln_b, mb16, 1);
  run_moe(ts_t_ffn, 1, mb16, guide, h16, wb16, rwb, gbb, ybuf, stream);
  add_t_kernel<<<addb, 256, 0, stream>>>(xts, ybuf);
  // spatial attention
  ln_kernel<<<lnb, 256, 0, stream>>>(xts, tss_ln_g, tss_ln_b, mb16, 0);
  cvt(tss_qkv_w, wb16, QKVW, stream);
  launch_gemm(mb16, wb16, tss_qkv_b, qkvh, MTOK, 3 * DIMc, DIMc, stream);
  attn_kernel<<<dim3(G_SP, HEADS), 256, 0, stream>>>(qkvh, mb16, S_SP, fcos, fsin, 0);
  cvt(tss_out_w, wb16, PRJW, stream);
  launch_gemm(mb16, wb16, tss_out_b, ybuf, MTOK, DIMc, DIMc, stream);
  add_kernel<<<addb, 256, 0, stream>>>(xts, ybuf, ntot);
  // spatial MoE
  ln_kernel<<<lnb, 256, 0, stream>>>(xts, tss_ffn_ln_g, tss_ffn_ln_b, mb16, 0);
  run_moe(ts_s_ffn, 0, mb16, guide, h16, wb16, rwb, gbb, ybuf, stream);
  add_kernel<<<addb, 256, 0, stream>>>(xts, ybuf, ntot);

  // ================= fuse =================
  fuse_kernel<<<ceil_div(MTOK, 128), 128, 0, stream>>>(xst, xts, fuse_w, fuse_b,
                                                       (float*)d_out);
}
